// CGCNNet_28046136443437
// MI455X (gfx1250) — matrix operand order repaired and checked by reading, not compiled, not run on hardware
//
#include <hip/hip_runtime.h>
#include <math.h>

// ---------------------------------------------------------------------------
// CGCNN forward on gfx1250 (MI455X), compile-only target.
//
// Factorized CGConv:  z@W = x[dst]@W_d + x[src]@W_s + e@W_e
//   * node projections (N x fout, WMMA f16 GEMM) -> fp16 tables (L2 resident)
//   * fused edge kernel: WMMA(e_tile @ W_e) + gather node terms + sigmoid*softplus
//     + global_atomic_add_f32 scatter into agg[dst]
//   * combine: relu(agg + x@Wt + bt) -> fp16 for next layer
//   * BN batch-stats reduction + tiny classifier
// ---------------------------------------------------------------------------

typedef __attribute__((ext_vector_type(16))) _Float16 v16h;
typedef __attribute__((ext_vector_type(8)))  _Float16 v8h;
typedef __attribute__((ext_vector_type(8)))  float    v8f;

#define NN      20000
#define EE      320000
#define FNODE   92
#define FEDGE   41
#define KP_X    96      // 92 padded to mult of 32
#define KP_E    64      // 41 padded to mult of 32
#define KP_H    128
#define HH      128
#define OO      256
#define CC      10
#define BN_EPS  1e-5f

#define DEV __device__ __forceinline__

// ---- WMMA fragment loaders (CDNA5 16-bit layouts, CDNA5 ISA) ----
// A (16x32, MxK), row-major src with leading dim lda (halves):
//   lane L: row M = L&15, kgrp = L>>4
//   halves 0..7  -> K = k0 + kgrp*8 + i        (contiguous 16B)
//   halves 8..15 -> K = k0 + 16 + kgrp*8 + i-8 (contiguous 16B)
DEV v16h load_frag_a(const _Float16* __restrict__ A, int lda, int row0, int k0, int lane) {
    int r  = row0 + (lane & 15);
    int kg = (lane >> 4) & 1;
    const _Float16* p = A + (size_t)r * lda + k0 + kg * 8;
    v8h lo = *reinterpret_cast<const v8h*>(p);
    v8h hi = *reinterpret_cast<const v8h*>(p + 16);
    v16h out;
#pragma unroll
    for (int i = 0; i < 8; ++i) { out[i] = lo[i]; out[i + 8] = hi[i]; }
    return out;
}

// B (32x16, KxN) taken from a pre-transposed weight pack Bt[fout][Kpad]:
//   lane L: col N = L&15, kgrp = L>>4 ; halves 0..7 -> K = k0 + kgrp*8 + i,
//   halves 8..15 -> K = k0 + 16 + kgrp*8 + i-8  => two 16B loads.
DEV v16h load_frag_b(const _Float16* __restrict__ Bt, int ldb, int c0, int k0, int lane) {
    int c  = c0 + (lane & 15);
    int kg = (lane >> 4) & 1;
    const _Float16* p = Bt + (size_t)c * ldb + k0 + kg * 8;
    v8h lo = *reinterpret_cast<const v8h*>(p);
    v8h hi = *reinterpret_cast<const v8h*>(p + 16);
    v16h out;
#pragma unroll
    for (int i = 0; i < 8; ++i) { out[i] = lo[i]; out[i + 8] = hi[i]; }
    return out;
}

DEV v8f wmma_f16(v16h a, v16h b, v8f c) {
    return __builtin_amdgcn_wmma_f32_16x16x32_f16(false, a, false, b, (short)0, c, false, false);
}

// ---- elementwise helpers ---------------------------------------------------
__global__ __launch_bounds__(256) void k_zero(float* __restrict__ p, int n) {
    int i = blockIdx.x * 256 + threadIdx.x;
    if (i < n) p[i] = 0.f;
}

// fp32 [rows x cols] -> fp16 [rows x colsPad] zero-padded
__global__ __launch_bounds__(256) void k_convert_pad(const float* __restrict__ in, int rows,
                                                     int cols, int colsPad,
                                                     _Float16* __restrict__ out) {
    int gid = blockIdx.x * 256 + threadIdx.x;
    if (gid >= rows * colsPad) return;
    int i = gid / colsPad, k = gid - i * colsPad;
    out[gid] = (k < cols) ? (_Float16)in[(size_t)i * cols + k] : (_Float16)0.f;
}

// W[zdim x fout] rows [row0, row0+nrows) -> transposed fp16 pack [fout x Kpad]
__global__ __launch_bounds__(256) void k_pack_w(const float* __restrict__ W, int ldW, int row0,
                                                int nrows, int ncols, int Kpad,
                                                _Float16* __restrict__ out) {
    int gid = blockIdx.x * 256 + threadIdx.x;
    if (gid >= ncols * Kpad) return;
    int c = gid / Kpad, k = gid - c * Kpad;
    out[gid] = (k < nrows) ? (_Float16)W[(size_t)(row0 + k) * ldW + c] : (_Float16)0.f;
}

// ---- node-projection GEMM: C[N x M] = A_f16[N x Kpad] @ Bt^T (+bias) -------
// block = 256 threads = 8 waves; each wave -> one 16x16 tile; block tile 16x128
__global__ __launch_bounds__(256) void k_gemm(const _Float16* __restrict__ A,
                                              const _Float16* __restrict__ Bt,
                                              const float* __restrict__ bias,
                                              float* __restrict__ C32,
                                              _Float16* __restrict__ C16,
                                              int Kpad, int M) {
    int lane = threadIdx.x & 31;
    int wave = threadIdx.x >> 5;
    int row0 = blockIdx.x * 16;
    int c0   = blockIdx.y * 128 + wave * 16;
    v8f acc = {};
    for (int k0 = 0; k0 < Kpad; k0 += 32) {
        v16h a = load_frag_a(A, Kpad, row0, k0, lane);
        v16h b = load_frag_b(Bt, Kpad, c0, k0, lane);
        acc = wmma_f16(a, b, acc);
    }
    // C/D layout: VGPR r -> M = r + 8*(lane>=16), N = lane&15
    int col = c0 + (lane & 15);
    int rb  = row0 + ((lane >> 4) << 3);
    float bv = bias ? bias[col] : 0.f;
#pragma unroll
    for (int r = 0; r < 8; ++r) {
        float v = acc[r] + bv;
        size_t idx = (size_t)(rb + r) * M + col;
        if (C32) C32[idx] = v;
        if (C16) C16[idx] = (_Float16)v;
    }
}

// ---- fused edge kernel: WMMA(e @ We) + gathers + gate + scatter-add --------
// block = 128 threads = 4 waves; wave tile = 16 edges x 16 features
__global__ __launch_bounds__(128) void k_edge(const _Float16* __restrict__ e16,
                                              const _Float16* __restrict__ WfE,
                                              const _Float16* __restrict__ WsE,
                                              const float* __restrict__ bf,
                                              const float* __restrict__ bs,
                                              const _Float16* __restrict__ Pfd,
                                              const _Float16* __restrict__ Pfs,
                                              const _Float16* __restrict__ Psd,
                                              const _Float16* __restrict__ Pss,
                                              const int* __restrict__ src,
                                              const int* __restrict__ dst,
                                              float* __restrict__ agg, int fout) {
    int lane = threadIdx.x & 31;
    int wave = threadIdx.x >> 5;
    int e0 = blockIdx.x * 16;
    int c0 = blockIdx.y * 64 + wave * 16;
    v8f aF = {}, aS = {};
#pragma unroll
    for (int k0 = 0; k0 < KP_E; k0 += 32) {
        v16h a  = load_frag_a(e16, KP_E, e0, k0, lane);
        v16h b1 = load_frag_b(WfE, KP_E, c0, k0, lane);
        aF = wmma_f16(a, b1, aF);
        v16h b2 = load_frag_b(WsE, KP_E, c0, k0, lane);
        aS = wmma_f16(a, b2, aS);
    }
    int col = c0 + (lane & 15);
    int mb  = (lane >> 4) << 3;
    float bfv = bf[col], bsv = bs[col];
#pragma unroll
    for (int r = 0; r < 8; ++r) {
        int e = e0 + mb + r;
        int d = dst[e], s = src[e];
        size_t di = (size_t)d * fout + col;
        size_t si = (size_t)s * fout + col;
        float f  = aF[r] + bfv + (float)Pfd[di] + (float)Pfs[si];
        float g  = 1.f / (1.f + expf(-f));
        float sv = aS[r] + bsv + (float)Psd[di] + (float)Pss[si];
        float sp = (sv > 20.f) ? sv : log1pf(expf(sv));
        atomicAdd(&agg[di], g * sp);   // global_atomic_add_f32, L2-resident
    }
}

// ---- h = (relu)(agg + T); optional fp32 and fp16 outputs -------------------
__global__ __launch_bounds__(256) void k_combine(const float* __restrict__ agg,
                                                 const float* __restrict__ T,
                                                 float* __restrict__ h32,
                                                 _Float16* __restrict__ h16,
                                                 int n, int relu) {
    int i = blockIdx.x * 256 + threadIdx.x;
    if (i >= n) return;
    float v = agg[i] + T[i];
    if (relu) v = fmaxf(v, 0.f);
    if (h32) h32[i] = v;
    if (h16) h16[i] = (_Float16)v;
}

// ---- BatchNorm batch statistics -> scale/shift per feature -----------------
__global__ __launch_bounds__(256) void k_bn(const float* __restrict__ h,
                                            const float* __restrict__ gamma,
                                            const float* __restrict__ beta,
                                            float* __restrict__ scale,
                                            float* __restrict__ shift) {
    __shared__ float s1[256], s2[256];
    int c = blockIdx.x, t = threadIdx.x;
    float a = 0.f, b = 0.f;
    for (int i = t; i < NN; i += 256) {
        float v = h[(size_t)i * OO + c];
        a += v; b += v * v;
    }
    s1[t] = a; s2[t] = b;
    __syncthreads();
    for (int s = 128; s > 0; s >>= 1) {
        if (t < s) { s1[t] += s1[t + s]; s2[t] += s2[t + s]; }
        __syncthreads();
    }
    if (t == 0) {
        float mean = s1[0] / (float)NN;
        float var  = s2[0] / (float)NN - mean * mean;
        float sc   = gamma[c] * rsqrtf(var + BN_EPS);
        scale[c] = sc;
        shift[c] = beta[c] - mean * sc;
    }
}

// ---- classifier: out[i][c] = bc[c] + sum_k bn(h[i][k]) * Wc[k][c] ----------
__global__ __launch_bounds__(256) void k_cls(const float* __restrict__ h,
                                             const float* __restrict__ scale,
                                             const float* __restrict__ shift,
                                             const float* __restrict__ Wc,
                                             const float* __restrict__ bc,
                                             float* __restrict__ out, int total) {
    int gid = blockIdx.x * 256 + threadIdx.x;
    if (gid >= total) return;
    int i = gid / CC, c = gid - i * CC;
    const float* hr = h + (size_t)i * OO;
    float acc = bc[c];
    for (int k = 0; k < OO; ++k)
        acc += (hr[k] * scale[k] + shift[k]) * Wc[k * CC + c];
    out[gid] = acc;
}

// ---------------------------------------------------------------------------
extern "C" void kernel_launch(void* const* d_in, const int* in_sizes, int n_in,
                              void* d_out, int out_size, void* d_ws, size_t ws_size,
                              hipStream_t stream) {
    (void)in_sizes; (void)n_in; (void)out_size; (void)ws_size;
    // setup_inputs() insertion order: x, edge_index, edge_attr,
    //   conv1{Wf,bf,Ws,bs,Wt,bt}, conv2{...}, conv3{...}, bn{gamma,beta}, cls{W,b}
    const float* x   = (const float*)d_in[0];
    const int*   ei  = (const int*)d_in[1];      // [2][E]: src = ei, dst = ei+E
    const float* ea  = (const float*)d_in[2];
    const float* Wf[3] = {(const float*)d_in[3],  (const float*)d_in[9],  (const float*)d_in[15]};
    const float* bf[3] = {(const float*)d_in[4],  (const float*)d_in[10], (const float*)d_in[16]};
    const float* Ws[3] = {(const float*)d_in[5],  (const float*)d_in[11], (const float*)d_in[17]};
    const float* bs[3] = {(const float*)d_in[6],  (const float*)d_in[12], (const float*)d_in[18]};
    const float* Wt[3] = {(const float*)d_in[7],  (const float*)d_in[13], (const float*)d_in[19]};
    const float* bt[3] = {(const float*)d_in[8],  (const float*)d_in[14], (const float*)d_in[20]};
    const float* gamma = (const float*)d_in[21];
    const float* beta  = (const float*)d_in[22];
    const float* Wc    = (const float*)d_in[23];
    const float* bc    = (const float*)d_in[24];
    float* out = (float*)d_out;

    // ---- workspace carve-out (256B aligned) ----
    char*  ws  = (char*)d_ws;
    size_t off = 0;
    auto take = [&](size_t bytes) -> char* {
        char* p = ws + off;
        off = (off + bytes + 255) & ~(size_t)255;
        return p;
    };
    _Float16* x16  = (_Float16*)take((size_t)NN * KP_X * 2);
    _Float16* e16  = (_Float16*)take((size_t)EE * KP_E * 2);
    _Float16* h16a = (_Float16*)take((size_t)NN * KP_H * 2);
    _Float16* h16b = (_Float16*)take((size_t)NN * KP_H * 2);
    _Float16* Pfd  = (_Float16*)take((size_t)NN * OO * 2);
    _Float16* Pfs  = (_Float16*)take((size_t)NN * OO * 2);
    _Float16* Psd  = (_Float16*)take((size_t)NN * OO * 2);
    _Float16* Pss  = (_Float16*)take((size_t)NN * OO * 2);
    float*    Tb   = (float*)take((size_t)NN * OO * 4);
    float*    agg  = (float*)take((size_t)NN * OO * 4);
    float*    h3   = (float*)take((size_t)NN * OO * 4);
    _Float16* WfD  = (_Float16*)take((size_t)OO * KP_H * 2);
    _Float16* WfS  = (_Float16*)take((size_t)OO * KP_H * 2);
    _Float16* WsD  = (_Float16*)take((size_t)OO * KP_H * 2);
    _Float16* WsS  = (_Float16*)take((size_t)OO * KP_H * 2);
    _Float16* WTp  = (_Float16*)take((size_t)OO * KP_H * 2);
    _Float16* WfE  = (_Float16*)take((size_t)OO * KP_E * 2);
    _Float16* WsE  = (_Float16*)take((size_t)OO * KP_E * 2);
    float*    bnsc = (float*)take(OO * 4);
    float*    bnsh = (float*)take(OO * 4);

    auto cdiv = [](int a, int b) { return (a + b - 1) / b; };

    // ---- stage 0: fp32 -> fp16 padded conversions ----
    k_convert_pad<<<cdiv(NN * KP_X, 256), 256, 0, stream>>>(x, NN, FNODE, KP_X, x16);
    k_convert_pad<<<cdiv(EE * KP_E, 256), 256, 0, stream>>>(ea, EE, FEDGE, KP_E, e16);

    auto run_layer = [&](const _Float16* A16, int Kpad, int fin, int fout, int li,
                         _Float16* h16out, float* h32out, int relu) {
        int zoff = 2 * fin;
        int pw = fout * Kpad, pe = fout * KP_E;
        // pack weight slices transposed into fp16
        k_pack_w<<<cdiv(pw, 256), 256, 0, stream>>>(Wf[li], fout, 0,    fin,   fout, Kpad, WfD);
        k_pack_w<<<cdiv(pw, 256), 256, 0, stream>>>(Wf[li], fout, fin,  fin,   fout, Kpad, WfS);
        k_pack_w<<<cdiv(pw, 256), 256, 0, stream>>>(Ws[li], fout, 0,    fin,   fout, Kpad, WsD);
        k_pack_w<<<cdiv(pw, 256), 256, 0, stream>>>(Ws[li], fout, fin,  fin,   fout, Kpad, WsS);
        k_pack_w<<<cdiv(pw, 256), 256, 0, stream>>>(Wt[li], fout, 0,    fin,   fout, Kpad, WTp);
        k_pack_w<<<cdiv(pe, 256), 256, 0, stream>>>(Wf[li], fout, zoff, FEDGE, fout, KP_E, WfE);
        k_pack_w<<<cdiv(pe, 256), 256, 0, stream>>>(Ws[li], fout, zoff, FEDGE, fout, KP_E, WsE);
        // node projections (WMMA GEMMs)
        dim3 gg(NN / 16, fout / 128, 1);
        k_gemm<<<gg, 256, 0, stream>>>(A16, WfD, nullptr, nullptr, Pfd, Kpad, fout);
        k_gemm<<<gg, 256, 0, stream>>>(A16, WfS, nullptr, nullptr, Pfs, Kpad, fout);
        k_gemm<<<gg, 256, 0, stream>>>(A16, WsD, nullptr, nullptr, Psd, Kpad, fout);
        k_gemm<<<gg, 256, 0, stream>>>(A16, WsS, nullptr, nullptr, Pss, Kpad, fout);
        k_gemm<<<gg, 256, 0, stream>>>(A16, WTp, bt[li], Tb, nullptr, Kpad, fout);
        // scatter accumulator
        int na = NN * fout;
        k_zero<<<cdiv(na, 256), 256, 0, stream>>>(agg, na);
        // fused edge GEMM + gate + scatter
        dim3 ge(EE / 16, fout / 64, 1);
        k_edge<<<ge, 128, 0, stream>>>(e16, WfE, WsE, bf[li], bs[li],
                                       Pfd, Pfs, Psd, Pss, ei, ei + EE, agg, fout);
        // combine + activation + next-layer fp16
        k_combine<<<cdiv(na, 256), 256, 0, stream>>>(agg, Tb, h32out, h16out, na, relu);
    };

    run_layer(x16,  KP_X, FNODE, HH, 0, h16a, nullptr, 1);
    run_layer(h16a, KP_H, HH,    HH, 1, h16b, nullptr, 1);
    run_layer(h16b, KP_H, HH,    OO, 2, nullptr, h3,   0);

    k_bn<<<OO, 256, 0, stream>>>(h3, gamma, beta, bnsc, bnsh);
    int tc = NN * CC;
    k_cls<<<cdiv(tc, 256), 256, 0, stream>>>(h3, bnsc, bnsh, Wc, bc, out, tc);
}
